// DecoderBlock_39719857553753
// MI455X (gfx1250) — matrix operand order repaired and checked by reading, not compiled, not run on hardware
//
#include <hip/hip_runtime.h>
#include <hip/hip_bf16.h>
#include <math.h>

#define B_   2
#define TQ_  2048
#define TK_  2048
#define D_   1024
#define H_   16
#define HD_  64
#define MLP_ 4096

typedef __attribute__((ext_vector_type(16))) __bf16 v16bf;
typedef __attribute__((ext_vector_type(8)))  __bf16 v8bf;
typedef __attribute__((ext_vector_type(8)))  float  v8f;
typedef unsigned short u16;
typedef int v4i __attribute__((vector_size(16)));
typedef v4i __attribute__((address_space(1)))* g128p;   // global int4*
typedef v4i __attribute__((address_space(3)))* l128p;   // LDS int4*

#if defined(__has_builtin)
#if __has_builtin(__builtin_amdgcn_global_load_async_to_lds_b128)
#define ASYNC_LDS 1
#endif
#endif

__device__ __forceinline__ u16 f2bf(float f) {
  unsigned u = __float_as_uint(f);
  u += 0x7FFFu + ((u >> 16) & 1u);      // round-to-nearest-even
  return (u16)(u >> 16);
}

__device__ __forceinline__ v8f v8f_zero() {
  v8f z;
#pragma unroll
  for (int i = 0; i < 8; ++i) z[i] = 0.f;
  return z;
}

__device__ __forceinline__ v16bf cat16(v8bf lo, v8bf hi) {
  return __builtin_shufflevector(lo, hi, 0,1,2,3,4,5,6,7,8,9,10,11,12,13,14,15);
}

// 16-byte global -> LDS stage; async (ASYNCcnt) path when the toolchain has it
__device__ __forceinline__ void stage16B(const u16* g, u16* l) {
#ifdef ASYNC_LDS
  __builtin_amdgcn_global_load_async_to_lds_b128((g128p)g, (l128p)l, 0, 0);
#else
  *(v8bf*)l = *(const v8bf*)g;
#endif
}
__device__ __forceinline__ void stage_wait() {
#ifdef ASYNC_LDS
  asm volatile("s_wait_asynccnt 0x0" ::: "memory");
#endif
}

// ---------------------------------------------------------------- converts
__global__ void cvt_f32_bf16(const float* __restrict__ src, u16* __restrict__ dst, long n) {
  long i = (long)blockIdx.x * blockDim.x + threadIdx.x;
  long stride = (long)gridDim.x * blockDim.x;
  for (; i < n; i += stride) dst[i] = f2bf(src[i]);
}

// W[K,N] f32  ->  Wt[N,K] bf16 (one-time; lets GEMM stage B tiles with b128 copies)
__global__ __launch_bounds__(256)
void cvt_w_transpose(const float* __restrict__ src, u16* __restrict__ dst, int K, int N) {
  __shared__ float tile[32][33];
  const int nb = blockIdx.x * 32;
  const int kb = blockIdx.y * 32;
  const int tx = threadIdx.x & 31;
  const int ty = threadIdx.x >> 5;   // 8 rows per pass
#pragma unroll
  for (int i = 0; i < 32; i += 8)
    tile[ty + i][tx] = src[(size_t)(kb + ty + i) * N + nb + tx];
  __syncthreads();
#pragma unroll
  for (int i = 0; i < 32; i += 8)
    dst[(size_t)(nb + ty + i) * K + kb + tx] = f2bf(tile[tx][ty + i]);
}

// ---------------------------------------------------------------- layernorm -> bf16
__global__ __launch_bounds__(256)
void ln_bf16_kernel(const float* __restrict__ x, const float* __restrict__ g,
                    const float* __restrict__ bta, u16* __restrict__ out) {
  const int row = blockIdx.x;
  const float* xr = x + (size_t)row * D_;
  __shared__ float s1[256];
  __shared__ float s2[256];
  const int t = threadIdx.x;
  float v0[4];
  float s = 0.f, sq = 0.f;
#pragma unroll
  for (int i = 0; i < 4; ++i) {
    float v = xr[t + 256 * i];
    v0[i] = v; s += v; sq += v * v;
  }
  s1[t] = s; s2[t] = sq; __syncthreads();
  for (int o = 128; o > 0; o >>= 1) {
    if (t < o) { s1[t] += s1[t + o]; s2[t] += s2[t + o]; }
    __syncthreads();
  }
  const float mean = s1[0] * (1.f / D_);
  const float var  = s2[0] * (1.f / D_) - mean * mean;
  const float r    = rsqrtf(var + 1e-5f);
  u16* orow = out + (size_t)row * D_;
#pragma unroll
  for (int i = 0; i < 4; ++i) {
    int c = t + 256 * i;
    orow[c] = f2bf((v0[i] - mean) * r * g[c] + bta[c]);
  }
}

// ---------------------------------------------------------------- GEMM (bf16 in, f32 accum)
// C[M,N] = A[M,K] @ Wt[N,K]^T + bias ; optional GELU / residual ; f32 and/or bf16 out.
// Block: 256 threads (8 wave32), tile 128x128, K-step 64 (16 WMMA per barrier pair).
template<bool GELU, bool RESID, bool OUTF, bool OUTB>
__global__ __launch_bounds__(256)
void gemm_bf16(const u16* __restrict__ A, const u16* __restrict__ Wt,
               const float* __restrict__ bias, const float* __restrict__ resid,
               float* __restrict__ outF, u16* __restrict__ outB,
               int M, int N, int K) {
  __shared__ __align__(16) u16 As[128 * 72];   // [row][k] pad->72
  __shared__ __align__(16) u16 Bs[128 * 72];   // [col][k] pad->72

  const int t        = threadIdx.x;
  const int lane     = t & 31;
  const int wave     = t >> 5;
  const int wm       = wave & 3;    // 4 slabs of 32 rows
  const int wn       = wave >> 2;   // 2 slabs of 64 cols
  const int lane16   = lane & 15;
  const int laneHalf = lane >> 4;
  const int mBase    = blockIdx.y * 128;
  const int nBase    = blockIdx.x * 128;

  v8f acc[2][4];
#pragma unroll
  for (int i = 0; i < 2; ++i)
#pragma unroll
    for (int j = 0; j < 4; ++j) acc[i][j] = v8f_zero();

  const int srow = t >> 1;          // staged row/col 0..127
  const int shalf = (t & 1) * 32;   // k half

  for (int k0 = 0; k0 < K; k0 += 64) {
    {
      const u16* ap = A  + (size_t)(mBase + srow) * K + k0 + shalf;
      const u16* bp = Wt + (size_t)(nBase + srow) * K + k0 + shalf;
      if (k0 + 64 < K) { __builtin_prefetch(ap + 64, 0, 3); __builtin_prefetch(bp + 64, 0, 3); }
#pragma unroll
      for (int c = 0; c < 4; ++c) {
        stage16B(ap + c * 8, &As[srow * 72 + shalf + c * 8]);
        stage16B(bp + c * 8, &Bs[srow * 72 + shalf + c * 8]);
      }
    }
    stage_wait();
    __syncthreads();

#pragma unroll
    for (int kk = 0; kk < 64; kk += 32) {
      v16bf af[2], bw[4];
#pragma unroll
      for (int mi = 0; mi < 2; ++mi) {
        int row = wm * 32 + mi * 16 + lane16;
        v8bf lo = *(const v8bf*)(&As[row * 72 + kk + laneHalf * 8]);
        v8bf hi = *(const v8bf*)(&As[row * 72 + kk + laneHalf * 8 + 16]);
        af[mi] = cat16(lo, hi);
      }
#pragma unroll
      for (int ni = 0; ni < 4; ++ni) {
        int col = wn * 64 + ni * 16 + lane16;
        v8bf lo = *(const v8bf*)(&Bs[col * 72 + kk + laneHalf * 8]);
        v8bf hi = *(const v8bf*)(&Bs[col * 72 + kk + laneHalf * 8 + 16]);
        bw[ni] = cat16(lo, hi);
      }
#pragma unroll
      for (int mi = 0; mi < 2; ++mi)
#pragma unroll
        for (int ni = 0; ni < 4; ++ni)
          acc[mi][ni] = __builtin_amdgcn_wmma_f32_16x16x32_bf16(
              false, af[mi], false, bw[ni], (short)0, acc[mi][ni], false, false);
    }
    __syncthreads();
  }

  // epilogue
#pragma unroll
  for (int mi = 0; mi < 2; ++mi) {
#pragma unroll
    for (int ni = 0; ni < 4; ++ni) {
      int col = nBase + wn * 64 + ni * 16 + lane16;
      float bv = bias[col];
#pragma unroll
      for (int v = 0; v < 8; ++v) {
        int row = mBase + wm * 32 + mi * 16 + v + laneHalf * 8;
        float xv = acc[mi][ni][v] + bv;
        if (GELU) xv = 0.5f * xv * (1.f + erff(xv * 0.70710678f));
        if (RESID) xv += resid[(size_t)row * N + col];
        if (OUTF) outF[(size_t)row * N + col] = xv;
        if (OUTB) outB[(size_t)row * N + col] = f2bf(xv);
      }
    }
  }
}

// ---------------------------------------------------------------- flash attention
// Block: 128 q rows (8 waves x 16 rows), 64-key tiles, HD=64, bf16 in/out, f32 softmax.
template<bool CAUSAL>
__global__ __launch_bounds__(256)
void flash_attn(const u16* __restrict__ Qb, int ldq,
                const u16* __restrict__ Kb, int ldk,
                const u16* __restrict__ Vb, int ldv,
                u16* __restrict__ Ob, int Tk) {
  __shared__ __align__(16) u16 Qs [128 * 72];     // [qrow][hd]
  __shared__ __align__(16) u16 Ks [64 * 72];      // [key][hd]
  __shared__ __align__(16) u16 Vts[64 * 72];      // transposed: [hd][key]
  __shared__ __align__(16) u16 Ps [8 * 16 * 72];  // per-wave P staging [wrow][key]

  const int t        = threadIdx.x;
  const int lane     = t & 31;
  const int wave     = t >> 5;
  const int lane16   = lane & 15;
  const int laneHalf = lane >> 4;

  const int qTile = blockIdx.x * 128;
  const int h     = blockIdx.y;
  const int b     = blockIdx.z;

  const u16* Q  = Qb + (size_t)b * TQ_ * ldq + h * HD_;
  const u16* Kp = Kb + (size_t)b * Tk  * ldk + h * HD_;
  const u16* Vp = Vb + (size_t)b * Tk  * ldv + h * HD_;

  // stage Q tile once
  {
    int row  = t >> 1;
    int half = (t & 1) * 32;
    const u16* qp = Q + (size_t)(qTile + row) * ldq + half;
#pragma unroll
    for (int c = 0; c < 4; ++c)
      stage16B(qp + c * 8, &Qs[row * 72 + half + c * 8]);
  }

  float mstat[8], lstat[8];
  v8f Oacc[4];
#pragma unroll
  for (int v = 0; v < 8; ++v) { mstat[v] = -1e30f; lstat[v] = 0.f; }
#pragma unroll
  for (int ni = 0; ni < 4; ++ni) Oacc[ni] = v8f_zero();

  const float scale = 0.125f;  // HD^-0.5
  const int kLimit = CAUSAL ? ((qTile + 128 < Tk) ? qTile + 128 : Tk) : Tk;
  stage_wait();
  __syncthreads();

  for (int kb = 0; kb < kLimit; kb += 64) {
    // stage K tile [key][hd]
    {
      int row = t >> 2;
      int ch  = (t & 3) * 16;
      const u16* kp = Kp + (size_t)(kb + row) * ldk + ch;
      stage16B(kp,     &Ks[row * 72 + ch]);
      stage16B(kp + 8, &Ks[row * 72 + ch + 8]);
    }
    // stage V tile transposed [hd][key] (element scatter: must stay manual)
#pragma unroll
    for (int i = 0; i < 2; ++i) {
      int idx = t * 2 + i;
      int key = idx >> 3;
      int hc  = (idx & 7) * 8;
      const u16* vp = Vp + (size_t)(kb + key) * ldv + hc;
      v8bf vv = *(const v8bf*)vp;
#pragma unroll
      for (int j = 0; j < 8; ++j) Vts[(hc + j) * 72 + key] = ((u16*)&vv)[j];
    }
    stage_wait();
    __syncthreads();

    // S = Q * K^T  (16 q rows per wave x 64 keys)
    v8f S[4];
#pragma unroll
    for (int ni = 0; ni < 4; ++ni) S[ni] = v8f_zero();
    const int qrow = wave * 16 + lane16;
#pragma unroll
    for (int ks = 0; ks < 2; ++ks) {
      int kof = ks * 32;
      v8bf qlo = *(const v8bf*)(&Qs[qrow * 72 + kof + laneHalf * 8]);
      v8bf qhi = *(const v8bf*)(&Qs[qrow * 72 + kof + laneHalf * 8 + 16]);
      v16bf aq = cat16(qlo, qhi);
#pragma unroll
      for (int ni = 0; ni < 4; ++ni) {
        int key = ni * 16 + lane16;
        v8bf bl = *(const v8bf*)(&Ks[key * 72 + kof + laneHalf * 8]);
        v8bf bh = *(const v8bf*)(&Ks[key * 72 + kof + laneHalf * 8 + 16]);
        S[ni] = __builtin_amdgcn_wmma_f32_16x16x32_bf16(
            false, aq, false, cat16(bl, bh), (short)0, S[ni], false, false);
      }
    }

    // online softmax (rows live in 16-lane halves; VGPR v -> row v (+8 for hi half))
    float tmax[8];
#pragma unroll
    for (int v = 0; v < 8; ++v) {
      float mx = -1e30f;
#pragma unroll
      for (int ni = 0; ni < 4; ++ni) {
        float s = S[ni][v] * scale;
        if (CAUSAL) {
          int qi = qTile + wave * 16 + v + laneHalf * 8;
          int ki = kb + ni * 16 + lane16;
          if (ki > qi) s = -1e30f;
        }
        S[ni][v] = s;
        mx = fmaxf(mx, s);
      }
#pragma unroll
      for (int o = 1; o < 16; o <<= 1) mx = fmaxf(mx, __shfl_xor(mx, o, 32));
      tmax[v] = mx;
    }
#pragma unroll
    for (int v = 0; v < 8; ++v) {
      float mnew  = fmaxf(mstat[v], tmax[v]);
      float alpha = __expf(mstat[v] - mnew);
      mstat[v] = mnew;
      float rs = 0.f;
#pragma unroll
      for (int ni = 0; ni < 4; ++ni) {
        float p = __expf(S[ni][v] - mnew);
        rs += p;
        Ps[(wave * 16 + v + laneHalf * 8) * 72 + ni * 16 + lane16] = f2bf(p);
      }
#pragma unroll
      for (int o = 1; o < 16; o <<= 1) rs += __shfl_xor(rs, o, 32);
      lstat[v] = lstat[v] * alpha + rs;
#pragma unroll
      for (int ni = 0; ni < 4; ++ni) Oacc[ni][v] *= alpha;
    }

    // wave-private LDS RAW: DS ops are in-order per wave; fence the compiler + hw
    asm volatile("s_wait_dscnt 0x0" ::: "memory");

    // O += P * V
#pragma unroll
    for (int ks = 0; ks < 2; ++ks) {
      int kof = ks * 32;
      const u16* pr = &Ps[(wave * 16 + lane16) * 72 + kof];
      v8bf plo = *(const v8bf*)(pr + laneHalf * 8);
      v8bf phi = *(const v8bf*)(pr + laneHalf * 8 + 16);
      v16bf ap = cat16(plo, phi);
#pragma unroll
      for (int ni = 0; ni < 4; ++ni) {
        int hd = ni * 16 + lane16;
        v8bf bl = *(const v8bf*)(&Vts[hd * 72 + kof + laneHalf * 8]);
        v8bf bh = *(const v8bf*)(&Vts[hd * 72 + kof + laneHalf * 8 + 16]);
        Oacc[ni] = __builtin_amdgcn_wmma_f32_16x16x32_bf16(
            false, ap, false, cat16(bl, bh), (short)0, Oacc[ni], false, false);
      }
    }
    __syncthreads();
  }

  // finalize: O /= l, write bf16 (heads interleaved into [*, D])
#pragma unroll
  for (int v = 0; v < 8; ++v) {
    float inv = 1.f / lstat[v];
#pragma unroll
    for (int ni = 0; ni < 4; ++ni) {
      int row = qTile + wave * 16 + v + laneHalf * 8;
      int col = h * HD_ + ni * 16 + lane16;
      Ob[(size_t)(b * TQ_ + row) * D_ + col] = f2bf(Oacc[ni][v] * inv);
    }
  }
}

// ---------------------------------------------------------------- host
extern "C" void kernel_launch(void* const* d_in, const int* in_sizes, int n_in,
                              void* d_out, int out_size, void* d_ws, size_t ws_size,
                              hipStream_t stream) {
  const float* x      = (const float*)d_in[0];
  const float* enc    = (const float*)d_in[1];
  // d_in[2] = causal mask (tril) -- handled analytically
  const float* ln1g = (const float*)d_in[3];
  const float* ln1b = (const float*)d_in[4];
  const float* ln2g = (const float*)d_in[5];
  const float* ln2b = (const float*)d_in[6];
  const float* ln3g = (const float*)d_in[7];
  const float* ln3b = (const float*)d_in[8];
  const float* qkv_w  = (const float*)d_in[9];
  const float* qkv_b  = (const float*)d_in[10];
  const float* proj_w = (const float*)d_in[11];
  const float* proj_b = (const float*)d_in[12];
  const float* q_w    = (const float*)d_in[13];
  const float* q_b    = (const float*)d_in[14];
  const float* kv_w   = (const float*)d_in[15];
  const float* kv_b   = (const float*)d_in[16];
  const float* co_w   = (const float*)d_in[17];
  const float* co_b   = (const float*)d_in[18];
  const float* mlp1_w = (const float*)d_in[19];
  const float* mlp1_b = (const float*)d_in[20];
  const float* mlp2_w = (const float*)d_in[21];
  const float* mlp2_b = (const float*)d_in[22];
  float* out = (float*)d_out;

  char* ws = (char*)d_ws;
  size_t off = 0;
  auto alloc = [&](size_t bytes) -> void* {
    void* p = ws + off;
    off += (bytes + 255) & ~(size_t)255;
    return p;
  };
  const size_t DD = (size_t)D_ * D_;
  const size_t BT = (size_t)B_ * TQ_;

  u16* w_qkv  = (u16*)alloc(3 * DD * 2);   // all weights stored transposed [N][K] bf16
  u16* w_proj = (u16*)alloc(DD * 2);
  u16* w_q    = (u16*)alloc(DD * 2);
  u16* w_kv   = (u16*)alloc(2 * DD * 2);
  u16* w_co   = (u16*)alloc(DD * 2);
  u16* w_m1   = (u16*)alloc(4 * DD * 2);
  u16* w_m2   = (u16*)alloc(4 * DD * 2);
  u16* qkvb   = (u16*)alloc(BT * 3 * D_ * 2);  // reused (with attnb) as MLP hidden
  u16* attnb  = (u16*)alloc(BT * D_ * 2);
  u16* bufA   = (u16*)alloc(BT * D_ * 2);      // ln1/ln2/ln3 outputs (sequential)
  u16* kvb    = (u16*)alloc(BT * 2 * D_ * 2);
  u16* qcb    = (u16*)alloc(BT * D_ * 2);
  u16* encb   = (u16*)alloc(BT * D_ * 2);
  u16* crossb = (u16*)alloc(BT * D_ * 2);
  float* x1   = (float*)alloc(BT * D_ * 4);
  float* x2   = (float*)alloc(BT * D_ * 4);
  u16* m1b    = qkvb;  // 24MB + 8MB contiguous == exactly B*TQ*MLP bf16

  // weight conversion: f32 [K,N] -> bf16 [N,K] (transposed once, L2-resident after)
  cvt_w_transpose<<<dim3(3 * D_ / 32, D_ / 32), 256, 0, stream>>>(qkv_w,  w_qkv,  D_, 3 * D_);
  cvt_w_transpose<<<dim3(D_ / 32,     D_ / 32), 256, 0, stream>>>(proj_w, w_proj, D_, D_);
  cvt_w_transpose<<<dim3(D_ / 32,     D_ / 32), 256, 0, stream>>>(q_w,    w_q,    D_, D_);
  cvt_w_transpose<<<dim3(2 * D_ / 32, D_ / 32), 256, 0, stream>>>(kv_w,   w_kv,   D_, 2 * D_);
  cvt_w_transpose<<<dim3(D_ / 32,     D_ / 32), 256, 0, stream>>>(co_w,   w_co,   D_, D_);
  cvt_w_transpose<<<dim3(MLP_ / 32,   D_ / 32), 256, 0, stream>>>(mlp1_w, w_m1,   D_, MLP_);
  cvt_w_transpose<<<dim3(D_ / 32,   MLP_ / 32), 256, 0, stream>>>(mlp2_w, w_m2,   MLP_, D_);
  cvt_f32_bf16<<<2048, 256, 0, stream>>>(enc, encb, (long)(BT * D_));

  const int M = (int)BT;  // 4096

  // ---- self attention ----
  ln_bf16_kernel<<<M, 256, 0, stream>>>(x, ln1g, ln1b, bufA);
  gemm_bf16<false, false, false, true><<<dim3(3 * D_ / 128, M / 128), 256, 0, stream>>>(
      bufA, w_qkv, qkv_b, nullptr, nullptr, qkvb, M, 3 * D_, D_);
  flash_attn<true><<<dim3(TQ_ / 128, H_, B_), 256, 0, stream>>>(
      qkvb, 3 * D_, qkvb + D_, 3 * D_, qkvb + 2 * D_, 3 * D_, attnb, TQ_);
  gemm_bf16<false, true, true, false><<<dim3(D_ / 128, M / 128), 256, 0, stream>>>(
      attnb, w_proj, proj_b, x, x1, nullptr, M, D_, D_);

  // ---- cross attention ----
  ln_bf16_kernel<<<M, 256, 0, stream>>>(x1, ln2g, ln2b, bufA);
  gemm_bf16<false, false, false, true><<<dim3(D_ / 128, M / 128), 256, 0, stream>>>(
      bufA, w_q, q_b, nullptr, nullptr, qcb, M, D_, D_);
  gemm_bf16<false, false, false, true><<<dim3(2 * D_ / 128, M / 128), 256, 0, stream>>>(
      encb, w_kv, kv_b, nullptr, nullptr, kvb, M, 2 * D_, D_);
  flash_attn<false><<<dim3(TQ_ / 128, H_, B_), 256, 0, stream>>>(
      qcb, D_, kvb, 2 * D_, kvb + D_, 2 * D_, crossb, TK_);
  gemm_bf16<false, true, true, false><<<dim3(D_ / 128, M / 128), 256, 0, stream>>>(
      crossb, w_co, co_b, x1, x2, nullptr, M, D_, D_);

  // ---- MLP ----
  ln_bf16_kernel<<<M, 256, 0, stream>>>(x2, ln3g, ln3b, bufA);
  gemm_bf16<true, false, false, true><<<dim3(MLP_ / 128, M / 128), 256, 0, stream>>>(
      bufA, w_m1, mlp1_b, nullptr, nullptr, m1b, M, MLP_, D_);
  gemm_bf16<false, true, true, false><<<dim3(D_ / 128, M / 128), 256, 0, stream>>>(
      m1b, w_m2, mlp2_b, x2, out, nullptr, M, D_, MLP_);
}
